// DeepHydro_85177791414302
// MI455X (gfx1250) — compile-verified, matrix operand order checked by reading, not run on hardware
//
#include <hip/hip_runtime.h>

typedef __attribute__((ext_vector_type(16))) _Float16 v16h;
typedef __attribute__((ext_vector_type(8)))  _Float16 v8h;
typedef __attribute__((ext_vector_type(8)))  float    v8f;

// ---------------- workspace layout ----------------
// f16 packed-weight region (units: halves)
constexpr size_t OW_CELL1  = 0;                     // [ug.w1|rg.w1] Kp288 x Np256
constexpr size_t OW_CELLN1 = OW_CELL1  + 288*256;   // ns.w1 288x128
constexpr size_t OW_CELLU2 = OW_CELLN1 + 288*128;   // 128x128
constexpr size_t OW_CELLR2 = OW_CELLU2 + 128*128;
constexpr size_t OW_CELLN2 = OW_CELLR2 + 128*128;   // 128x256
constexpr size_t OW_Z1     = OW_CELLN2 + 128*256;   // [zug.w1|zrg.w1] 256x256
constexpr size_t OW_ZN1    = OW_Z1     + 256*256;   // 256x128
constexpr size_t OW_ZU2    = OW_ZN1    + 256*128;   // 128x64
constexpr size_t OW_ZR2    = OW_ZU2    + 128*64;
constexpr size_t OW_ZN2    = OW_ZR2    + 128*64;    // 128x112
constexpr size_t OW_ZO1    = OW_ZN2    + 128*112;   // 64x64
constexpr size_t OW_ZO2    = OW_ZO1    + 64*64;
constexpr size_t OW_CNF1   = OW_ZO2    + 64*64;     // 64x64
constexpr size_t OW_CNF2   = OW_CNF1   + 64*64;
constexpr size_t OW_FWIH   = OW_CNF2   + 64*64;     // 192x384
constexpr size_t OW_FWHH   = OW_FWIH   + 192*384;   // 128x384
constexpr size_t OW_BWIH   = OW_FWHH   + 128*384;
constexpr size_t OW_BWHH   = OW_BWIH   + 192*384;
constexpr size_t OW_F11    = OW_BWHH   + 128*384;   // 32x32
constexpr size_t OW_F12    = OW_F11    + 32*32;     // 32x128
constexpr size_t OW_TO1    = OW_F12    + 32*128;    // 128x128
constexpr size_t OW_TO2    = OW_TO1    + 128*128;
constexpr size_t OW_F21    = OW_TO2    + 128*128;   // 32x32
constexpr size_t OW_F22    = OW_F21    + 32*32;     // 32x128
constexpr size_t OW_LL1    = OW_F22    + 32*128;
constexpr size_t OW_LL2    = OW_LL1    + 128*128;
constexpr size_t OW_O1     = OW_LL2    + 128*128;   // 576x128
constexpr size_t OW_O2     = OW_O1     + 576*128;   // 128x32
constexpr size_t OW_O3     = OW_O2     + 128*32;    // 32x16
constexpr size_t H_END     = OW_O3     + 32*16;
constexpr size_t BIAS_BASE = ((H_END*2 + 255)/256)*256;   // byte offset

// f32 padded-bias region (units: floats, relative to BIAS_BASE)
constexpr size_t OB_CELL1  = 0;               // 256
constexpr size_t OB_CELLN1 = OB_CELL1  + 256; // 128
constexpr size_t OB_CELLU2 = OB_CELLN1 + 128;
constexpr size_t OB_CELLR2 = OB_CELLU2 + 128;
constexpr size_t OB_CELLN2 = OB_CELLR2 + 128; // 256
constexpr size_t OB_Z1     = OB_CELLN2 + 256; // 256
constexpr size_t OB_ZN1    = OB_Z1     + 256; // 128
constexpr size_t OB_ZU2    = OB_ZN1    + 128; // 64
constexpr size_t OB_ZR2    = OB_ZU2    + 64;
constexpr size_t OB_ZN2    = OB_ZR2    + 64;  // 112
constexpr size_t OB_ZO1    = OB_ZN2    + 112; // 64
constexpr size_t OB_ZO2    = OB_ZO1    + 64;
constexpr size_t OB_CNF1   = OB_ZO2    + 64;  // 64
constexpr size_t OB_CNF2   = OB_CNF1   + 64;  // 64
constexpr size_t OB_FWIH   = OB_CNF2   + 64;  // 384
constexpr size_t OB_FWHH   = OB_FWIH   + 384;
constexpr size_t OB_BWIH   = OB_FWHH   + 384;
constexpr size_t OB_BWHH   = OB_BWIH   + 384;
constexpr size_t OB_F11    = OB_BWHH   + 384; // 32
constexpr size_t OB_F12    = OB_F11    + 32;  // 128
constexpr size_t OB_TO1    = OB_F12    + 128;
constexpr size_t OB_TO2    = OB_TO1    + 128;
constexpr size_t OB_F21    = OB_TO2    + 128; // 32
constexpr size_t OB_F22    = OB_F21    + 32;  // 128
constexpr size_t OB_LL1    = OB_F22    + 128;
constexpr size_t OB_LL2    = OB_LL1    + 128;
constexpr size_t OB_O1     = OB_LL2    + 128; // 128
constexpr size_t OB_O2     = OB_O1     + 128; // 32
constexpr size_t OB_O3     = OB_O2     + 32;  // 16
constexpr size_t B_END     = OB_O3     + 16;

constexpr size_t SEQ_BASE   = ((BIAS_BASE + B_END*4 + 255)/256)*256;   // bytes
constexpr size_t SEQ_HALVES = (size_t)168*512*192;                     // f16 [t][b][192]
constexpr size_t F32_BASE   = ((SEQ_BASE + SEQ_HALVES*2 + 255)/256)*256;
constexpr size_t OF_ZFIN  = 0;                 // 512*64 f32
constexpr size_t OF_ZSFIN = OF_ZFIN  + 512*64;
constexpr size_t OF_FWL   = OF_ZSFIN + 512*64; // 512*128
constexpr size_t OF_BWL   = OF_FWL   + 512*128;
constexpr size_t OF_ZK    = OF_BWL   + 512*128; // 512*64
constexpr size_t WS_NEEDED = F32_BASE + (OF_ZK + 512*64)*4;

// ---------------- WMMA helpers ----------------
__device__ __forceinline__ v8f wmma16(v16h a, v16h b, v8f c) {
  return __builtin_amdgcn_wmma_f32_16x16x32_f16(false, a, false, b, (short)0, c, false, false);
}

// A fragment from a row-contiguous f16 buffer (16 rows, stride in halves, mult of 8)
__device__ __forceinline__ v16h load_afrag(const _Float16* base, int stride, int kt, int lane) {
  const _Float16* p = base + (size_t)(lane & 15) * stride + kt * 32 + ((lane & 16) ? 8 : 0);
  v8h lo = *(const v8h*)p;
  v8h hi = *(const v8h*)(p + 16);
  v16h r;
#pragma unroll
  for (int i = 0; i < 8; ++i) { r[i] = lo[i]; r[i + 8] = hi[i]; }
  return r;
}

template <int ACT>
__device__ __forceinline__ float act_apply(float x, float ap) {
  if (ACT == 1) return tanhf(x);
  if (ACT == 2) return 1.f / (1.f + __expf(-x));
  if (ACT == 3) return x > 0.f ? x : (__expf(x) - 1.f);   // elu
  if (ACT == 4) return fmaxf(x, 0.f);                     // relu
  if (ACT == 5) return x >= 0.f ? x : ap * x;             // prelu
  return x;
}

// C += A(16xKT*32) * W(KT*32 x NT*16), bias+act, f32 out
template <int ACT>
__device__ __forceinline__ void gemm_f32(const _Float16* A, int aStride,
    const _Float16* W, int KT, int NT, const float* bias,
    float* out, int outStride, int wave, int nwaves, int lane, float ap = 0.f) {
  for (int nt = wave; nt < NT; nt += nwaves) {
    v8f c = {};
    for (int kt = 0; kt < KT; ++kt) {
      v16h a = load_afrag(A, aStride, kt, lane);
      v16h b = *(const v16h*)(W + (((size_t)kt * NT + nt) * 32 + (size_t)lane) * 16);
      c = wmma16(a, b, c);
    }
    int col = nt * 16 + (lane & 15);
    float bv = bias ? bias[col] : 0.f;
    int rb = (lane & 16) ? 8 : 0;
#pragma unroll
    for (int v = 0; v < 8; ++v)
      out[(size_t)(rb + v) * outStride + col] = act_apply<ACT>(c[v] + bv, ap);
  }
}

// same but f16 output (feeds the next GEMM's A operand)
template <int ACT>
__device__ __forceinline__ void gemm_f16(const _Float16* A, int aStride,
    const _Float16* W, int KT, int NT, const float* bias,
    _Float16* out, int outStride, int wave, int nwaves, int lane, float ap = 0.f) {
  for (int nt = wave; nt < NT; nt += nwaves) {
    v8f c = {};
    for (int kt = 0; kt < KT; ++kt) {
      v16h a = load_afrag(A, aStride, kt, lane);
      v16h b = *(const v16h*)(W + (((size_t)kt * NT + nt) * 32 + (size_t)lane) * 16);
      c = wmma16(a, b, c);
    }
    int col = nt * 16 + (lane & 15);
    float bv = bias ? bias[col] : 0.f;
    int rb = (lane & 16) ? 8 : 0;
#pragma unroll
    for (int v = 0; v < 8; ++v)
      out[(size_t)(rb + v) * outStride + col] = (_Float16)act_apply<ACT>(c[v] + bv, ap);
  }
}

// ---------------- deterministic RNG (stand-in for fixed jax PRNG draws) ----------------
__device__ __forceinline__ unsigned hash_u32(unsigned x) {
  x ^= x >> 16; x *= 0x7feb352dU; x ^= x >> 15; x *= 0x846ca68bU; x ^= x >> 16;
  return x;
}
__device__ __forceinline__ float gauss_at(unsigned idx, unsigned seed) {
  unsigned u1 = hash_u32(idx * 2u + 0x9e3779b9U * seed + 1u);
  unsigned u2 = hash_u32(idx * 2u + 0x9e3779b9U * seed + 2u);
  float f1 = ((float)u1 + 1.0f) * (1.0f / 4294967296.0f);
  float f2 = (float)u2 * (1.0f / 4294967296.0f);
  return sqrtf(-2.f * __logf(f1)) * __cosf(6.28318530718f * f2);
}
__device__ __forceinline__ float eps_val(int t, int b, int j) {
  return gauss_at((unsigned)((t * 512 + b) * 50 + j), 42u);
}
__device__ __forceinline__ float rad_val(int b, int j) {
  return (hash_u32((unsigned)(b * 50 + j) ^ 0xabcdef01u) & 1u) ? 1.f : -1.f;
}

// ---------------- weight / bias pre-pack ----------------
__global__ void pack_w_kernel(const float* __restrict__ src, _Float16* __restrict__ dst,
                              int K, int N, int NTtot, int ntStart, int ntCount, int nOff, int total) {
  int idx = blockIdx.x * blockDim.x + threadIdx.x;
  if (idx >= total) return;
  int i = idx & 15;
  int lane = (idx >> 4) & 31;
  int f = idx >> 9;
  int ntl = f % ntCount;
  int kt = f / ntCount;
  int nt = ntStart + ntl;
  int ncol = nt * 16 + (lane & 15);
  int nrel = ncol - nOff;
  int klocal = (lane < 16) ? ((i < 8) ? i : 8 + i) : ((i < 8) ? 8 + i : 16 + i);
  int k = kt * 32 + klocal;
  float v = (k < K && nrel >= 0 && nrel < N) ? src[(size_t)k * N + nrel] : 0.f;
  dst[(((size_t)(kt * NTtot + nt)) * 32 + lane) * 16 + i] = (_Float16)v;
}

__global__ void pack_b_kernel(const float* __restrict__ src, float* __restrict__ dst,
                              int N, int nOff, int count) {
  int j = blockIdx.x * blockDim.x + threadIdx.x;
  if (j >= count) return;
  dst[nOff + j] = (j < N) ? src[j] : 0.f;
}

__global__ void zero_kl_kernel(float* out) { if (threadIdx.x == 0) out[512] = 0.f; }

// ---------------- kernel 1: recurrent cell/zcell/zode over 168 steps ----------------
__global__ __launch_bounds__(512) void rnn_kernel(const float* __restrict__ s_inp,
                                                  const float* __restrict__ flow_x,
                                                  char* __restrict__ ws) {
  const _Float16* WH = (const _Float16*)ws;
  const float* BB = (const float*)(ws + BIAS_BASE);
  _Float16* seq = (_Float16*)(ws + SEQ_BASE);
  float* f32s = (float*)(ws + F32_BASE);
  float* zfin = f32s + OF_ZFIN;
  float* zsfin = f32s + OF_ZSFIN;

  __shared__ alignas(16) float sPy[16 * 132], sPs[16 * 132];
  __shared__ alignas(16) float sPzm[16 * 64], sPzs[16 * 64];
  __shared__ alignas(16) float sU[16 * 128], sR[16 * 128];
  __shared__ alignas(16) float sU2[16 * 64], sR2[16 * 64];
  __shared__ alignas(16) float sOut[16 * 392];
  __shared__ alignas(32) _Float16 sA[16 * 320];
  __shared__ alignas(32) _Float16 sH[16 * 392];

  const int tid = threadIdx.x, lane = tid & 31, wave = tid >> 5;
  const int b0 = blockIdx.x * 16;

  for (int i = tid; i < 16 * 132; i += 512) { sPy[i] = 0.f; sPs[i] = 0.f; }
  for (int i = tid; i < 16 * 64; i += 512) { sPzm[i] = 0.f; sPzs[i] = 0.f; }
  __syncthreads();

  for (int t = 0; t < 168; ++t) {
    // c = [py, ps, x]  (262 -> pad 288)
    for (int idx = tid; idx < 16 * 288; idx += 512) {
      int r = idx / 288, c = idx - r * 288;
      float v;
      if (c < 128) v = sPy[r * 132 + c];
      else if (c < 256) v = sPs[r * 132 + (c - 128)];
      else if (c < 259) v = s_inp[((size_t)(b0 + r) * 168 + t) * 3 + (c - 256)];
      else if (c < 262) v = flow_x[((size_t)(b0 + r) * 168 + t) * 3 + (c - 259)];
      else v = 0.f;
      sA[r * 320 + c] = (_Float16)v;
    }
    __syncthreads();
    // hidden for u,r gates (16x256) = tanh(c @ [ug.w1|rg.w1] + b)
    gemm_f16<1>(sA, 320, WH + OW_CELL1, 9, 16, BB + OB_CELL1, sH, 392, wave, 16, lane);
    __syncthreads();
    gemm_f32<2>(sH,       392, WH + OW_CELLU2, 4, 8, BB + OB_CELLU2, sU, 128, wave, 16, lane);
    gemm_f32<2>(sH + 128, 392, WH + OW_CELLR2, 4, 8, BB + OB_CELLR2, sR, 128, wave, 16, lane);
    __syncthreads();
    // c2 = [py*r, ps*r, x]
    for (int idx = tid; idx < 16 * 288; idx += 512) {
      int r = idx / 288, c = idx - r * 288;
      float v;
      if (c < 128) v = sPy[r * 132 + c] * sR[r * 128 + c];
      else if (c < 256) v = sPs[r * 132 + (c - 128)] * sR[r * 128 + (c - 128)];
      else if (c < 259) v = s_inp[((size_t)(b0 + r) * 168 + t) * 3 + (c - 256)];
      else if (c < 262) v = flow_x[((size_t)(b0 + r) * 168 + t) * 3 + (c - 259)];
      else v = 0.f;
      sA[r * 320 + c] = (_Float16)v;
    }
    __syncthreads();
    gemm_f16<1>(sA, 320, WH + OW_CELLN1, 9, 8, BB + OB_CELLN1, sH, 392, wave, 16, lane);
    __syncthreads();
    gemm_f32<0>(sH, 392, WH + OW_CELLN2, 4, 16, BB + OB_CELLN2, sOut, 392, wave, 16, lane);
    __syncthreads();
    // blend -> new y/std; emit h to seq
    for (int idx = tid; idx < 16 * 128; idx += 512) {
      int r = idx >> 7, c = idx & 127;
      float u = sU[r * 128 + c];
      float m = sOut[r * 392 + c];
      float s = fabsf(sOut[r * 392 + 128 + c]);
      float ny = (1.f - u) * m + u * sPy[r * 132 + c];
      float ns = fabsf((1.f - u) * s + u * sPs[r * 132 + c]);
      sPy[r * 132 + c] = ny;
      sPs[r * 132 + c] = ns;
      seq[((size_t)t * 512 + b0 + r) * 192 + c] = (_Float16)ny;
    }
    __syncthreads();
    // zcell input [pzm, pzs, h] (228 -> pad 256)
    for (int idx = tid; idx < 16 * 256; idx += 512) {
      int r = idx >> 8, c = idx & 255;
      float v;
      if (c < 50) v = sPzm[r * 64 + c];
      else if (c < 100) v = sPzs[r * 64 + (c - 50)];
      else if (c < 228) v = sPy[r * 132 + (c - 100)];
      else v = 0.f;
      sA[r * 320 + c] = (_Float16)v;
    }
    __syncthreads();
    gemm_f16<1>(sA, 320, WH + OW_Z1, 8, 16, BB + OB_Z1, sH, 392, wave, 16, lane);
    __syncthreads();
    gemm_f32<2>(sH,       392, WH + OW_ZU2, 4, 4, BB + OB_ZU2, sU2, 64, wave, 16, lane);
    gemm_f32<2>(sH + 128, 392, WH + OW_ZR2, 4, 4, BB + OB_ZR2, sR2, 64, wave, 16, lane);
    __syncthreads();
    for (int idx = tid; idx < 16 * 256; idx += 512) {
      int r = idx >> 8, c = idx & 255;
      float v;
      if (c < 50) v = sPzm[r * 64 + c] * sR2[r * 64 + c];
      else if (c < 100) v = sPzs[r * 64 + (c - 50)] * sR2[r * 64 + (c - 50)];
      else if (c < 228) v = sPy[r * 132 + (c - 100)];
      else v = 0.f;
      sA[r * 320 + c] = (_Float16)v;
    }
    __syncthreads();
    gemm_f16<1>(sA, 320, WH + OW_ZN1, 8, 8, BB + OB_ZN1, sH, 392, wave, 16, lane);
    __syncthreads();
    gemm_f32<0>(sH, 392, WH + OW_ZN2, 4, 7, BB + OB_ZN2, sOut, 392, wave, 16, lane);
    __syncthreads();
    for (int idx = tid; idx < 16 * 50; idx += 512) {
      int r = idx / 50, c = idx - r * 50;
      float u = sU2[r * 64 + c];
      float m = sOut[r * 392 + c];
      float s = fabsf(sOut[r * 392 + 50 + c]);
      float nz = (1.f - u) * m + u * sPzm[r * 64 + c];
      float nzs = fabsf((1.f - u) * s + u * sPzs[r * 64 + c]);
      sPzm[r * 64 + c] = nz;
      sPzs[r * 64 + c] = nzs;
    }
    __syncthreads();
    // zode: z += 1*f(z); z += 49*f(z)
    for (int es = 0; es < 2; ++es) {
      float coef = (es == 0) ? 1.f : 49.f;
      for (int idx = tid; idx < 16 * 64; idx += 512) {
        int r = idx >> 6, c = idx & 63;
        sA[r * 320 + c] = (_Float16)((c < 50) ? sPzm[r * 64 + c] : 0.f);
      }
      __syncthreads();
      gemm_f16<1>(sA, 320, WH + OW_ZO1, 2, 4, BB + OB_ZO1, sH, 392, wave, 16, lane);
      __syncthreads();
      gemm_f32<0>(sH, 392, WH + OW_ZO2, 2, 4, BB + OB_ZO2, sOut, 392, wave, 16, lane);
      __syncthreads();
      for (int idx = tid; idx < 16 * 50; idx += 512) {
        int r = idx / 50, c = idx - r * 50;
        sPzm[r * 64 + c] += coef * sOut[r * 392 + c];
      }
      __syncthreads();
    }
    // zh = z + eps*zs -> seq cols [128,192)
    for (int idx = tid; idx < 16 * 64; idx += 512) {
      int r = idx >> 6, c = idx & 63;
      float v = 0.f;
      if (c < 50) v = sPzm[r * 64 + c] + eps_val(t, b0 + r, c) * sPzs[r * 64 + c];
      seq[((size_t)t * 512 + b0 + r) * 192 + 128 + c] = (_Float16)v;
    }
    __syncthreads();
  }
  for (int idx = tid; idx < 16 * 64; idx += 512) {
    int r = idx >> 6, c = idx & 63;
    zfin[(size_t)(b0 + r) * 64 + c]  = (c < 50) ? sPzm[r * 64 + c] : 0.f;
    zsfin[(size_t)(b0 + r) * 64 + c] = (c < 50) ? sPzs[r * 64 + c] : 0.f;
  }
}

// ---------------- kernel 2: bidirectional GRU (only final states needed) ----------------
__global__ __launch_bounds__(512) void bigru_kernel(char* __restrict__ ws) {
  const _Float16* WH = (const _Float16*)ws;
  const float* BB = (const float*)(ws + BIAS_BASE);
  const _Float16* seq = (const _Float16*)(ws + SEQ_BASE);
  float* f32s = (float*)(ws + F32_BASE);
  const int dir = blockIdx.y;
  const _Float16* Wih = WH + (dir ? OW_BWIH : OW_FWIH);
  const _Float16* Whh = WH + (dir ? OW_BWHH : OW_FWHH);
  const float* Bih = BB + (dir ? OB_BWIH : OB_FWIH);
  const float* Bhh = BB + (dir ? OB_BWHH : OB_FWHH);
  float* outH = f32s + (dir ? OF_BWL : OF_FWL);

  __shared__ alignas(16) float sHst[16 * 132];
  __shared__ alignas(32) _Float16 sHh[16 * 136];
  __shared__ alignas(16) float sGi[16 * 392];
  __shared__ alignas(16) float sGh[16 * 392];

  const int tid = threadIdx.x, lane = tid & 31, wave = tid >> 5;
  const int b0 = blockIdx.x * 16;
  for (int i = tid; i < 16 * 132; i += 512) sHst[i] = 0.f;
  __syncthreads();

  for (int step = 0; step < 168; ++step) {
    int t = dir ? (167 - step) : step;
    const _Float16* Ax = seq + ((size_t)t * 512 + b0) * 192;
    gemm_f32<0>(Ax, 192, Wih, 6, 24, Bih, sGi, 392, wave, 16, lane);
    for (int i = tid; i < 16 * 128; i += 512) {
      int r = i >> 7, c = i & 127;
      sHh[r * 136 + c] = (_Float16)sHst[r * 132 + c];
    }
    __syncthreads();
    gemm_f32<0>(sHh, 136, Whh, 4, 24, Bhh, sGh, 392, wave, 16, lane);
    __syncthreads();
    for (int i = tid; i < 16 * 128; i += 512) {
      int r = i >> 7, c = i & 127;
      float ir = sGi[r * 392 + c], iz = sGi[r * 392 + 128 + c], in = sGi[r * 392 + 256 + c];
      float hr = sGh[r * 392 + c], hz = sGh[r * 392 + 128 + c], hn = sGh[r * 392 + 256 + c];
      float rr = 1.f / (1.f + __expf(-(ir + hr)));
      float zg = 1.f / (1.f + __expf(-(iz + hz)));
      float nn = tanhf(in + rr * hn);
      sHst[r * 132 + c] = (1.f - zg) * nn + zg * sHst[r * 132 + c];
    }
    __syncthreads();
  }
  for (int i = tid; i < 16 * 128; i += 512) {
    int r = i >> 7, c = i & 127;
    outH[(size_t)(b0 + r) * 128 + c] = sHst[r * 132 + c];
  }
}

// ---------------- kernel 3: CNF with Hutchinson trace + kl accumulation ----------------
__global__ __launch_bounds__(256) void cnf_kernel(char* __restrict__ ws, float* __restrict__ d_out) {
  const _Float16* WH = (const _Float16*)ws;
  const float* BB = (const float*)(ws + BIAS_BASE);
  float* f32s = (float*)(ws + F32_BASE);
  const float* zfin = f32s + OF_ZFIN;
  const float* zsfin = f32s + OF_ZSFIN;
  float* zk = f32s + OF_ZK;

  __shared__ alignas(16) float sZ[16 * 64], sZ0d[16 * 64], sZs[16 * 64], sZf[16 * 64];
  __shared__ alignas(16) float sDa[16 * 64], sPre[16 * 64], sF[16 * 64], sDf[16 * 64];
  __shared__ alignas(32) _Float16 sAh[16 * 72], sHh[16 * 72], sDh[16 * 72];
  __shared__ float sDlp[16], sC[16];

  const int tid = threadIdx.x, lane = tid & 31, wave = tid >> 5;
  const int b0 = blockIdx.x * 16;

  for (int i = tid; i < 16 * 64; i += 256) {
    int r = i >> 6, c = i & 63;
    float zf = zfin[(size_t)(b0 + r) * 64 + c];
    float zs = zsfin[(size_t)(b0 + r) * 64 + c];
    float z0v = (c < 50) ? zf + eps_val(168, b0 + r, c) * zs : 0.f;
    sZf[i] = zf; sZs[i] = zs; sZ0d[i] = z0v; sZ[i] = z0v;
    sAh[r * 72 + c] = (_Float16)((c < 50) ? rad_val(b0 + r, c) : 0.f);
  }
  if (tid < 16) sDlp[tid] = 0.f;
  __syncthreads();
  // da = [rad, 0] @ w1 (no bias; constant across steps)
  gemm_f32<0>(sAh, 72, WH + OW_CNF1, 2, 4, nullptr, sDa, 64, wave, 8, lane);
  __syncthreads();

  const float dtc = 1.f / 20.f;
  for (int n = 0; n < 20; ++n) {
    float tcur = (float)n * dtc;
    for (int i = tid; i < 16 * 64; i += 256) {
      int r = i >> 6, c = i & 63;
      float v = (c < 50) ? sZ[i] : ((c == 50) ? tcur : 0.f);
      sAh[r * 72 + c] = (_Float16)v;
    }
    __syncthreads();
    gemm_f32<0>(sAh, 72, WH + OW_CNF1, 2, 4, BB + OB_CNF1, sPre, 64, wave, 8, lane);
    __syncthreads();
    for (int i = tid; i < 16 * 64; i += 256) {
      int r = i >> 6, c = i & 63;
      float h = tanhf(sPre[i]);
      float dh = (1.f - h * h) * sDa[i];
      sHh[r * 72 + c] = (_Float16)h;
      sDh[r * 72 + c] = (_Float16)dh;
    }
    __syncthreads();
    gemm_f32<0>(sHh, 72, WH + OW_CNF2, 2, 4, BB + OB_CNF2, sF, 64, wave, 8, lane);
    gemm_f32<0>(sDh, 72, WH + OW_CNF2, 2, 4, nullptr, sDf, 64, wave, 8, lane);
    __syncthreads();
    for (int i = tid; i < 16 * 50; i += 256) {
      int r = i / 50, c = i - r * 50;
      sZ[r * 64 + c] += dtc * sF[r * 64 + c];
    }
    if (tid < 16) {
      float tr = 0.f;
      for (int j = 0; j < 50; ++j) tr += sDf[tid * 64 + j] * rad_val(b0 + tid, j);
      sDlp[tid] -= dtc * tr;
    }
    __syncthreads();
  }
  for (int i = tid; i < 16 * 64; i += 256) {
    int r = i >> 6, c = i & 63;
    zk[(size_t)(b0 + r) * 64 + c] = (c < 50) ? sZ[i] : 0.f;
  }
  if (tid < 16) {
    int r = tid;
    float lq = 0.f, lp = 0.f;
    for (int j = 0; j < 50; ++j) {
      float zs = sZs[r * 64 + j];
      float d = sZ0d[r * 64 + j] - sZf[r * 64 + j];
      lq += -0.5f * (__logf(zs) + d * d / zs);
      lp += -0.5f * sZ[r * 64 + j] * sZ[r * 64 + j];
    }
    sC[r] = lq - lp + sDlp[r];
  }
  __syncthreads();
  if (tid == 0) {
    float s = 0.f;
    for (int r = 0; r < 16; ++r) s += sC[r];
    atomicAdd(d_out + 512, s * (1.f / 512.f));
  }
}

// ---------------- kernel 4: temporal/flow features + RK4 ODEs + output head ----------------
__device__ void ode_eval(const float* in, float* out,
                         const _Float16* W1, const float* B1,
                         const _Float16* W2, const float* B2,
                         _Float16* sAh, _Float16* sHh, int tid, int lane, int wave) {
  for (int i = tid; i < 16 * 128; i += 256) {
    int r = i >> 7, c = i & 127;
    sAh[r * 136 + c] = (_Float16)in[r * 136 + c];
  }
  __syncthreads();
  gemm_f16<4>(sAh, 136, W1, 4, 8, B1, sHh, 136, wave, 8, lane);
  __syncthreads();
  gemm_f32<0>(sHh, 136, W2, 4, 8, B2, out, 136, wave, 8, lane);
  __syncthreads();
}

__device__ void rk4_ode(float* z, float* sIn, float* sK, float* sAcc,
                        const _Float16* W1, const float* B1,
                        const _Float16* W2, const float* B2,
                        _Float16* sAh, _Float16* sHh, int tid, int lane, int wave) {
  const float dt = 0.25f;
  for (int s = 0; s < 4; ++s) {
    for (int st = 0; st < 4; ++st) {
      float cc = (st == 0) ? 0.f : ((st == 3) ? dt : 0.5f * dt);
      for (int i = tid; i < 16 * 128; i += 256) {
        int r = i >> 7, c = i & 127;
        sIn[r * 136 + c] = z[r * 136 + c] + ((st == 0) ? 0.f : cc * sK[r * 136 + c]);
      }
      __syncthreads();
      ode_eval(sIn, sK, W1, B1, W2, B2, sAh, sHh, tid, lane, wave);
      float w = (st == 0 || st == 3) ? 1.f : 2.f;
      for (int i = tid; i < 16 * 128; i += 256) {
        int r = i >> 7, c = i & 127;
        if (st == 0) sAcc[r * 136 + c] = sK[r * 136 + c];
        else sAcc[r * 136 + c] += w * sK[r * 136 + c];
      }
      __syncthreads();
    }
    for (int i = tid; i < 16 * 128; i += 256) {
      int r = i >> 7, c = i & 127;
      z[r * 136 + c] += (dt / 6.f) * sAcc[r * 136 + c];
    }
    __syncthreads();
  }
}

__global__ __launch_bounds__(256) void head_kernel(char* __restrict__ ws, const int* __restrict__ T,
    const float* __restrict__ ew, const float* __restrict__ eh, const float* __restrict__ flow_y,
    const float* __restrict__ pa1, const float* __restrict__ pa2, const float* __restrict__ pa3,
    float* __restrict__ d_out) {
  const _Float16* WH = (const _Float16*)ws;
  const float* BB = (const float*)(ws + BIAS_BASE);
  float* f32s = (float*)(ws + F32_BASE);
  const float* fwl = f32s + OF_FWL;
  const float* bwl = f32s + OF_BWL;
  const float* zk = f32s + OF_ZK;

  __shared__ alignas(16) float sTf[16 * 136], sSt[16 * 136];
  __shared__ alignas(16) float sIn[16 * 136], sK[16 * 136], sAcc[16 * 136], sO[16 * 136];
  __shared__ alignas(32) _Float16 sAh[16 * 136], sHh[16 * 136];
  __shared__ alignas(32) _Float16 sFeat[16 * 576];

  const int tid = threadIdx.x, lane = tid & 31, wave = tid >> 5;
  const int b0 = blockIdx.x * 16;
  const float a1 = pa1[0], a2 = pa2[0], a3 = pa3[0];

  // ---- temporal features ----
  for (int i = tid; i < 16 * 32; i += 256) {
    int r = i >> 5, c = i & 31;
    int wk = T[(b0 + r) * 2 + 0];
    int hr = T[(b0 + r) * 2 + 1];
    float v = 0.f;
    if (c < 2) v = ew[wk * 2 + c];
    else if (c < 5) v = eh[hr * 3 + (c - 2)];
    else if (c == 5) v = (float)hr / 24.f;
    sAh[r * 136 + c] = (_Float16)v;
  }
  __syncthreads();
  gemm_f16<3>(sAh, 136, WH + OW_F11, 1, 2, BB + OB_F11, sHh, 136, wave, 8, lane);
  __syncthreads();
  gemm_f32<0>(sHh, 136, WH + OW_F12, 1, 8, BB + OB_F12, sSt, 136, wave, 8, lane);
  __syncthreads();
  rk4_ode(sSt, sIn, sK, sAcc, WH + OW_TO1, BB + OB_TO1, WH + OW_TO2, BB + OB_TO2,
          sAh, sHh, tid, lane, wave);
  for (int i = tid; i < 16 * 136; i += 256) sTf[i] = sSt[i];
  __syncthreads();

  // ---- water-flow features ----
  for (int i = tid; i < 16 * 32; i += 256) {
    int r = i >> 5, c = i & 31;
    float v = (c < 3) ? flow_y[(b0 + r) * 3 + c] : 0.f;
    sAh[r * 136 + c] = (_Float16)v;
  }
  __syncthreads();
  gemm_f16<3>(sAh, 136, WH + OW_F21, 1, 2, BB + OB_F21, sHh, 136, wave, 8, lane);
  __syncthreads();
  gemm_f32<0>(sHh, 136, WH + OW_F22, 1, 8, BB + OB_F22, sSt, 136, wave, 8, lane);
  __syncthreads();
  rk4_ode(sSt, sIn, sK, sAcc, WH + OW_LL1, BB + OB_LL1, WH + OW_LL2, BB + OB_LL2,
          sAh, sHh, tid, lane, wave);

  // ---- feat = [fw, bw, zk, tf, lf] (562 -> pad 576) ----
  for (int i = tid; i < 16 * 576; i += 256) {
    int r = i / 576, c = i - r * 576;
    float v;
    if (c < 128) v = fwl[(size_t)(b0 + r) * 128 + c];
    else if (c < 256) v = bwl[(size_t)(b0 + r) * 128 + (c - 128)];
    else if (c < 306) v = zk[(size_t)(b0 + r) * 64 + (c - 256)];
    else if (c < 434) v = sTf[r * 136 + (c - 306)];
    else if (c < 562) v = sSt[r * 136 + (c - 434)];
    else v = 0.f;
    sFeat[r * 576 + c] = (_Float16)v;
  }
  __syncthreads();
  gemm_f32<5>(sFeat, 576, WH + OW_O1, 18, 8, BB + OB_O1, sO, 136, wave, 8, lane, a1);
  __syncthreads();
  for (int i = tid; i < 16 * 128; i += 256) {
    int r = i >> 7, c = i & 127;
    sAh[r * 136 + c] = (_Float16)sO[r * 136 + c];
  }
  __syncthreads();
  gemm_f32<5>(sAh, 136, WH + OW_O2, 4, 2, BB + OB_O2, sO, 136, wave, 8, lane, a2);
  __syncthreads();
  for (int i = tid; i < 16 * 32; i += 256) {
    int r = i >> 5, c = i & 31;
    sAh[r * 136 + c] = (_Float16)sO[r * 136 + c];
  }
  __syncthreads();
  gemm_f32<5>(sAh, 136, WH + OW_O3, 1, 1, BB + OB_O3, sO, 136, wave, 8, lane, a3);
  __syncthreads();
  if (tid < 16) d_out[b0 + tid] = sO[tid * 136 + 0];
}

// ---------------- host launch ----------------
extern "C" void kernel_launch(void* const* d_in, const int* in_sizes, int n_in,
                              void* d_out, int out_size, void* d_ws, size_t ws_size,
                              hipStream_t stream) {
  (void)in_sizes; (void)n_in; (void)out_size;
  if (ws_size < WS_NEEDED) return;
  char* ws = (char*)d_ws;
  _Float16* wsH = (_Float16*)ws;
  float* wsB = (float*)(ws + BIAS_BASE);
  auto F = [&](int i) { return (const float*)d_in[i]; };

  auto packW = [&](int idx, int K, int N, int Kp, int NTtot, int ntStart, int ntCount,
                   int nOff, size_t dst) {
    int total = (Kp / 32) * ntCount * 512;
    pack_w_kernel<<<(total + 255) / 256, 256, 0, stream>>>(
        F(idx), wsH + dst, K, N, NTtot, ntStart, ntCount, nOff, total);
  };
  auto packB = [&](int idx, int N, int nOff, int count, size_t dst) {
    pack_b_kernel<<<(count + 63) / 64, 64, 0, stream>>>(F(idx), wsB + dst, N, nOff, count);
  };

  // cell (GRU_unit): ug 4..7, rg 8..11, ns 12..15
  packW(4, 262, 128, 288, 16, 0, 8, 0, OW_CELL1);
  packW(8, 262, 128, 288, 16, 8, 8, 128, OW_CELL1);
  packW(12, 262, 128, 288, 8, 0, 8, 0, OW_CELLN1);
  packW(6, 128, 128, 128, 8, 0, 8, 0, OW_CELLU2);
  packW(10, 128, 128, 128, 8, 0, 8, 0, OW_CELLR2);
  packW(14, 128, 256, 128, 16, 0, 16, 0, OW_CELLN2);
  // zcell: ug 16..19, rg 20..23, ns 24..27
  packW(16, 228, 100, 256, 16, 0, 8, 0, OW_Z1);
  packW(20, 228, 100, 256, 16, 8, 8, 128, OW_Z1);
  packW(24, 228, 100, 256, 8, 0, 8, 0, OW_ZN1);
  packW(18, 100, 50, 128, 4, 0, 4, 0, OW_ZU2);
  packW(22, 100, 50, 128, 4, 0, 4, 0, OW_ZR2);
  packW(26, 100, 100, 128, 7, 0, 7, 0, OW_ZN2);
  // zode 28..31, cnf 32..35
  packW(28, 50, 50, 64, 4, 0, 4, 0, OW_ZO1);
  packW(30, 50, 50, 64, 4, 0, 4, 0, OW_ZO2);
  packW(32, 51, 64, 64, 4, 0, 4, 0, OW_CNF1);
  packW(34, 64, 50, 64, 4, 0, 4, 0, OW_CNF2);
  // bigru fw 36..39, bw 40..43
  packW(36, 178, 384, 192, 24, 0, 24, 0, OW_FWIH);
  packW(37, 128, 384, 128, 24, 0, 24, 0, OW_FWHH);
  packW(40, 178, 384, 192, 24, 0, 24, 0, OW_BWIH);
  packW(41, 128, 384, 128, 24, 0, 24, 0, OW_BWHH);
  // fcn1 46..49, tode 50..53, fcn2 54..57, llode 58..61
  packW(46, 6, 32, 32, 2, 0, 2, 0, OW_F11);
  packW(48, 32, 128, 32, 8, 0, 8, 0, OW_F12);
  packW(50, 128, 128, 128, 8, 0, 8, 0, OW_TO1);
  packW(52, 128, 128, 128, 8, 0, 8, 0, OW_TO2);
  packW(54, 3, 32, 32, 2, 0, 2, 0, OW_F21);
  packW(56, 32, 128, 32, 8, 0, 8, 0, OW_F22);
  packW(58, 128, 128, 128, 8, 0, 8, 0, OW_LL1);
  packW(60, 128, 128, 128, 8, 0, 8, 0, OW_LL2);
  // out head 62..70 (w1,b1,a1,w2,b2,a2,w3,b3,a3)
  packW(62, 562, 128, 576, 8, 0, 8, 0, OW_O1);
  packW(65, 128, 32, 128, 2, 0, 2, 0, OW_O2);
  packW(68, 32, 1, 32, 1, 0, 1, 0, OW_O3);

  packB(5, 128, 0, 128, OB_CELL1);   packB(9, 128, 128, 128, OB_CELL1);
  packB(13, 128, 0, 128, OB_CELLN1);
  packB(7, 128, 0, 128, OB_CELLU2);  packB(11, 128, 0, 128, OB_CELLR2);
  packB(15, 256, 0, 256, OB_CELLN2);
  packB(17, 100, 0, 128, OB_Z1);     packB(21, 100, 128, 128, OB_Z1);
  packB(25, 100, 0, 128, OB_ZN1);
  packB(19, 50, 0, 64, OB_ZU2);      packB(23, 50, 0, 64, OB_ZR2);
  packB(27, 100, 0, 112, OB_ZN2);
  packB(29, 50, 0, 64, OB_ZO1);      packB(31, 50, 0, 64, OB_ZO2);
  packB(33, 64, 0, 64, OB_CNF1);     packB(35, 50, 0, 64, OB_CNF2);
  packB(38, 384, 0, 384, OB_FWIH);   packB(39, 384, 0, 384, OB_FWHH);
  packB(42, 384, 0, 384, OB_BWIH);   packB(43, 384, 0, 384, OB_BWHH);
  packB(47, 32, 0, 32, OB_F11);      packB(49, 128, 0, 128, OB_F12);
  packB(51, 128, 0, 128, OB_TO1);    packB(53, 128, 0, 128, OB_TO2);
  packB(55, 32, 0, 32, OB_F21);      packB(57, 128, 0, 128, OB_F22);
  packB(59, 128, 0, 128, OB_LL1);    packB(61, 128, 0, 128, OB_LL2);
  packB(63, 128, 0, 128, OB_O1);     packB(66, 32, 0, 32, OB_O2);
  packB(69, 1, 0, 16, OB_O3);

  zero_kl_kernel<<<1, 32, 0, stream>>>((float*)d_out);
  rnn_kernel<<<32, 512, 0, stream>>>(F(0), F(1), ws);
  bigru_kernel<<<dim3(32, 2), 512, 0, stream>>>(ws);
  cnf_kernel<<<32, 256, 0, stream>>>(ws, (float*)d_out);
  head_kernel<<<32, 256, 0, stream>>>(ws, (const int*)d_in[3], F(44), F(45), F(2),
                                      F(64), F(67), F(70), (float*)d_out);
}
